// StreamingTransformerLayer_73435350827142
// MI455X (gfx1250) — compile-verified, matrix operand order checked by reading, not run on hardware
//
#include <hip/hip_runtime.h>

// ---------------------------------------------------------------------------
// CDNA5 (gfx1250) transformer layer: bf16 WMMA GEMMs (ping-pong double-
// buffered, MT x NT register blocking, global_prefetch) + register-resident
// flash attention with a transposed-V operand. wave32 everywhere.
// ---------------------------------------------------------------------------

typedef __bf16 bf16_t;
typedef __attribute__((ext_vector_type(16))) __bf16 v16bf;
typedef __attribute__((ext_vector_type(8)))  float  v8f;

constexpr int Tt  = 8192;   // sequence length
constexpr int Dd  = 512;    // model dim
constexpr int Dff = 2048;   // ffn dim
constexpr int LD3 = 1536;   // qkv row stride

struct FragU { unsigned u[8]; };

__device__ __forceinline__ unsigned f32_to_bf16_bits(float f) {
  unsigned u = __builtin_bit_cast(unsigned, f);
  return (u + 0x7FFFu + ((u >> 16) & 1u)) >> 16;   // round-to-nearest-even
}
__device__ __forceinline__ bf16_t f32_to_bf16(float f) {
  unsigned short s = (unsigned short)f32_to_bf16_bits(f);
  return __builtin_bit_cast(bf16_t, s);
}
__device__ __forceinline__ v8f v8f_zero() {
  v8f z = {0.f, 0.f, 0.f, 0.f, 0.f, 0.f, 0.f, 0.f};
  return z;
}
__device__ __forceinline__ v8f wmma_bf16(v16bf a, v16bf b, v8f c) {
  // D = A(16x32) * B(32x16) + C, fp32 accumulate
  return __builtin_amdgcn_wmma_f32_16x16x32_bf16(false, a, false, b,
                                                 (short)0, c, false, false);
}

// A-fragment (and row-major-B fragment) for a 16x32 bf16 tile.
// ISA layout: lane -> row r = lane&15 ; koff = (lane>=16)?8:0 ;
//   VGPR v holds k = (v&4?16:0) + koff + 2*(v&3), +1  (contiguous pair).
// Per lane this is two contiguous 16B chunks -> 2x global_load_b128.
__device__ __forceinline__ v16bf load_frag_rowmajor(const bf16_t* __restrict__ p,
                                                    int ld) {
  int lane = threadIdx.x & 31;
  int r    = lane & 15;
  int koff = (lane >> 4) ? 8 : 0;
  const bf16_t* row = p + (size_t)r * ld;
  FragU f;
#pragma unroll
  for (int v = 0; v < 8; ++v) {
    int k = ((v & 4) ? 16 : 0) + koff + 2 * (v & 3);
    f.u[v] = *(const unsigned*)(row + k);
  }
  return __builtin_bit_cast(v16bf, f);
}

// ---------------------------------------------------------------------------
// elementwise / reduction / transpose kernels
// ---------------------------------------------------------------------------
__global__ void cvt_f32_bf16(const float* __restrict__ in, bf16_t* __restrict__ out,
                             int n) {
  int i = blockIdx.x * blockDim.x + threadIdx.x;
  if (i < n) out[i] = f32_to_bf16(in[i]);
}

// one wave per row; D multiple of 32
__global__ __launch_bounds__(256) void layernorm_bf16(
    const float* __restrict__ x, const float* __restrict__ g,
    const float* __restrict__ b, bf16_t* __restrict__ out, int T_, int D_) {
  int row = blockIdx.x * (blockDim.x >> 5) + (threadIdx.x >> 5);
  if (row >= T_) return;
  int lane = threadIdx.x & 31;
  const float* xr = x + (size_t)row * D_;
  float s = 0.f, ss = 0.f;
  for (int i = lane; i < D_; i += 32) { float v = xr[i]; s += v; ss += v * v; }
#pragma unroll
  for (int off = 16; off; off >>= 1) {
    s  += __shfl_xor(s,  off, 32);
    ss += __shfl_xor(ss, off, 32);
  }
  float mu  = s / D_;
  float var = ss / D_ - mu * mu;
  float rs  = rsqrtf(var + 1e-5f);
  bf16_t* orow = out + (size_t)row * D_;
  for (int i = lane; i < D_; i += 32)
    orow[i] = f32_to_bf16((xr[i] - mu) * rs * g[i] + b[i]);
}

// 32x32 LDS-tiled bf16 transpose: out[c][r] = in[r][c]
__global__ __launch_bounds__(256) void transpose_bf16(
    const bf16_t* __restrict__ in, int ldi,
    bf16_t* __restrict__ out, int ldo) {
  __shared__ bf16_t tile[32][33];
  int rb = blockIdx.y * 32, cb = blockIdx.x * 32;
  int tx = threadIdx.x & 31, ty = threadIdx.x >> 5;   // 8 row-groups
#pragma unroll
  for (int i = ty; i < 32; i += 8)
    tile[i][tx] = in[(size_t)(rb + i) * ldi + cb + tx];
  __syncthreads();
#pragma unroll
  for (int i = ty; i < 32; i += 8)
    out[(size_t)(cb + i) * ldo + rb + tx] = tile[tx][i];
}

// ---------------------------------------------------------------------------
// WMMA GEMM:  C[m,n] = epi( sum_k A[m,k] * B[n,k] )
//   A: [M,K] bf16 row-major (lda), B: [N,K] bf16 row-major (ldb)
//   MT x NT register blocking; ping-pong double buffering (K % 64 == 0
//   required) so no inter-buffer register copies; global_prefetch_b8 ahead.
//   EPI 0: store bf16    EPI 1: gelu(tanh) -> bf16
//   EPI 2: Cf[m,n] = resid[m,n] + scale[n] * acc   (fp32)
// ---------------------------------------------------------------------------
template <int MT, int NT, int EPI>
__global__ __launch_bounds__(256) void gemm_wmma(
    const bf16_t* __restrict__ A, int lda,
    const bf16_t* __restrict__ B, int ldb,
    int M, int N, int K,
    bf16_t* __restrict__ Cb, float* __restrict__ Cf, int ldc,
    const float* __restrict__ resid, const float* __restrict__ scale) {
  int wave   = blockIdx.x * (blockDim.x >> 5) + (threadIdx.x >> 5);
  int tilesN = N / (16 * NT);
  int tm     = wave / tilesN;
  int tn     = wave % tilesN;
  if (tm >= M / (16 * MT)) return;
  int m0 = tm * 16 * MT, n0 = tn * 16 * NT;
  int lane = threadIdx.x & 31;

  const bf16_t* Arow[MT];
  const bf16_t* Brow[NT];
#pragma unroll
  for (int mt = 0; mt < MT; ++mt) Arow[mt] = A + (size_t)(m0 + mt * 16) * lda;
#pragma unroll
  for (int t = 0; t < NT; ++t)    Brow[t]  = B + (size_t)(n0 + t * 16) * ldb;

  v8f acc[MT][NT];
#pragma unroll
  for (int mt = 0; mt < MT; ++mt)
#pragma unroll
    for (int t = 0; t < NT; ++t) acc[mt][t] = v8f_zero();

  v16bf a0[MT], b0[NT], a1[MT], b1[NT];
#pragma unroll
  for (int mt = 0; mt < MT; ++mt) a0[mt] = load_frag_rowmajor(Arow[mt], lda);
#pragma unroll
  for (int t = 0; t < NT; ++t)    b0[t]  = load_frag_rowmajor(Brow[t], ldb);

  for (int k0 = 0; k0 < K; k0 += 64) {        // K % 64 == 0
    // issue loads for the odd half (k0+32)
#pragma unroll
    for (int mt = 0; mt < MT; ++mt) a1[mt] = load_frag_rowmajor(Arow[mt] + k0 + 32, lda);
#pragma unroll
    for (int t = 0; t < NT; ++t)    b1[t]  = load_frag_rowmajor(Brow[t] + k0 + 32, ldb);
    if (k0 + 256 < K) {   // prefetch ahead (gfx1250 global_prefetch_b8)
      int ra = m0 + ((lane >> 4) << 4) + (lane & 15);               // 32 A rows
      __builtin_prefetch(A + (size_t)ra * lda + k0 + 256, 0, 1);
      __builtin_prefetch(B + (size_t)(n0 + lane) * ldb + k0 + 256, 0, 1);
      if (NT * 16 > 32)
        __builtin_prefetch(B + (size_t)(n0 + 32 + lane) * ldb + k0 + 256, 0, 1);
    }
    // even-half WMMAs
#pragma unroll
    for (int mt = 0; mt < MT; ++mt)
#pragma unroll
      for (int t = 0; t < NT; ++t)
        acc[mt][t] = wmma_bf16(a0[mt], b0[t], acc[mt][t]);
    // issue loads for the next even half (k0+64)
    if (k0 + 64 < K) {
#pragma unroll
      for (int mt = 0; mt < MT; ++mt) a0[mt] = load_frag_rowmajor(Arow[mt] + k0 + 64, lda);
#pragma unroll
      for (int t = 0; t < NT; ++t)    b0[t]  = load_frag_rowmajor(Brow[t] + k0 + 64, ldb);
    }
    // odd-half WMMAs
#pragma unroll
    for (int mt = 0; mt < MT; ++mt)
#pragma unroll
      for (int t = 0; t < NT; ++t)
        acc[mt][t] = wmma_bf16(a1[mt], b1[t], acc[mt][t]);
  }

  int nloc  = lane & 15;
  int mbase = (lane >> 4) ? 8 : 0;
#pragma unroll
  for (int mt = 0; mt < MT; ++mt) {
#pragma unroll
    for (int t = 0; t < NT; ++t) {
#pragma unroll
      for (int v = 0; v < 8; ++v) {
        int   m   = m0 + mt * 16 + mbase + v;
        int   n   = n0 + t * 16 + nloc;
        float val = acc[mt][t][v];
        if constexpr (EPI == 0) {
          Cb[(size_t)m * ldc + n] = f32_to_bf16(val);
        } else if constexpr (EPI == 1) {
          float u  = 0.7978845608028654f * (val + 0.044715f * val * val * val);
          float ge = 0.5f * val * (1.0f + tanhf(u));
          Cb[(size_t)m * ldc + n] = f32_to_bf16(ge);
        } else {
          Cf[(size_t)m * ldc + n] =
              resid[(size_t)m * ldc + n] + scale[n] * val;
        }
      }
    }
  }
}

// ---------------------------------------------------------------------------
// flash attention, causal, scores[i,j] = K[i]·Q[j] (reference convention).
// One wave: 16 output rows (i-block) x 16*NT V columns.  S^T tiles via WMMA
// (A = Q rows j, B = K rows i); the C-layout of S^T repacks directly into the
// A-fragment layout of P for the P@V WMMA (no LDS, no shuffles).  V is used
// pre-transposed (Vt[c][j]) so its fragments are b128-vectorizable.
// ---------------------------------------------------------------------------
template <int NT>
__global__ __launch_bounds__(256) void attn_flash(
    const bf16_t* __restrict__ qkv, const bf16_t* __restrict__ Vt,
    bf16_t* __restrict__ obuf, int T_) {
  constexpr int COLC = Dd / (16 * NT);
  int gwave = blockIdx.x * (blockDim.x >> 5) + (threadIdx.x >> 5);
  int nIB   = T_ / 16;
  int i_blk = gwave / COLC;
  int cch   = gwave % COLC;
  if (i_blk >= nIB) return;
  int i0 = i_blk * 16;
  int c0 = cch * 16 * NT;

  int lane  = threadIdx.x & 31;
  int nloc  = lane & 15;
  int mbase = (lane >> 4) ? 8 : 0;

  const bf16_t* Qb = qkv;
  const bf16_t* Kb = qkv + 512;

  v8f oacc[NT];
#pragma unroll
  for (int t = 0; t < NT; ++t) oacc[t] = v8f_zero();
  float mrun = -__builtin_inff();
  float lrun = 0.f;

  for (int j0 = 0; j0 <= i0; j0 += 32) {
    bool c1 = (j0 + 16 <= i0);
    const bf16_t* Krow  = Kb + (size_t)i0 * LD3;
    const bf16_t* Q0row = Qb + (size_t)j0 * LD3;
    const bf16_t* Q1row = Qb + (size_t)(j0 + 16) * LD3;

    v8f s0 = v8f_zero(), s1 = v8f_zero();
    // ping-pong double-buffered K-loop over the head dim (Dd % 64 == 0)
    v16bf kA  = load_frag_rowmajor(Krow, LD3);
    v16bf q0A = load_frag_rowmajor(Q0row, LD3);
    v16bf q1A;
    if (c1) q1A = load_frag_rowmajor(Q1row, LD3);
    for (int k0 = 0; k0 < Dd; k0 += 64) {
      v16bf kB  = load_frag_rowmajor(Krow + k0 + 32, LD3);
      v16bf q0B = load_frag_rowmajor(Q0row + k0 + 32, LD3);
      v16bf q1B;
      if (c1) q1B = load_frag_rowmajor(Q1row + k0 + 32, LD3);
      s0 = wmma_bf16(q0A, kA, s0);
      if (c1) s1 = wmma_bf16(q1A, kA, s1);
      if (k0 + 64 < Dd) {
        kA  = load_frag_rowmajor(Krow + k0 + 64, LD3);
        q0A = load_frag_rowmajor(Q0row + k0 + 64, LD3);
        if (c1) q1A = load_frag_rowmajor(Q1row + k0 + 64, LD3);
      }
      s0 = wmma_bf16(q0B, kB, s0);
      if (c1) s1 = wmma_bf16(q1B, kB, s1);
    }
    // causal mask on diagonal tiles: element (i = i0+nloc, j = jc+mbase+v)
    if (j0 == i0) {
#pragma unroll
      for (int v = 0; v < 8; ++v)
        if (j0 + mbase + v > i0 + nloc) s0[v] = -__builtin_inff();
    }
    if (c1 && (j0 + 16 == i0)) {
#pragma unroll
      for (int v = 0; v < 8; ++v)
        if (j0 + 16 + mbase + v > i0 + nloc) s1[v] = -__builtin_inff();
    }
    // online softmax: reduce over j (per-lane VGPRs + xor-16 across halves)
    float tmax = -__builtin_inff();
#pragma unroll
    for (int v = 0; v < 8; ++v) tmax = fmaxf(tmax, s0[v]);
    if (c1) {
#pragma unroll
      for (int v = 0; v < 8; ++v) tmax = fmaxf(tmax, s1[v]);
    }
    tmax = fmaxf(tmax, __shfl_xor(tmax, 16, 32));
    float mnew = fmaxf(mrun, tmax);
    float corr = __expf(mrun - mnew);
    float e0[8], e1[8];
    float sum = 0.f;
#pragma unroll
    for (int v = 0; v < 8; ++v) { e0[v] = __expf(s0[v] - mnew); sum += e0[v]; }
    if (c1) {
#pragma unroll
      for (int v = 0; v < 8; ++v) { e1[v] = __expf(s1[v] - mnew); sum += e1[v]; }
    } else {
#pragma unroll
      for (int v = 0; v < 8; ++v) e1[v] = 0.f;
    }
    sum += __shfl_xor(sum, 16, 32);
    lrun = lrun * corr + sum;
    mrun = mnew;
    // rescale o accumulators: stats live at lane = i&15, o rows at mbase+v
    float cv[8];
#pragma unroll
    for (int v = 0; v < 8; ++v) cv[v] = __shfl(corr, mbase + v, 32);
#pragma unroll
    for (int t = 0; t < NT; ++t)
#pragma unroll
      for (int v = 0; v < 8; ++v) oacc[t][v] *= cv[v];
    // repack P (S^T C-layout -> A-fragment layout, same lane-half split)
    FragU pf;
#pragma unroll
    for (int v = 0; v < 4; ++v) {
      pf.u[v]     = f32_to_bf16_bits(e0[2 * v]) | (f32_to_bf16_bits(e0[2 * v + 1]) << 16);
      pf.u[4 + v] = f32_to_bf16_bits(e1[2 * v]) | (f32_to_bf16_bits(e1[2 * v + 1]) << 16);
    }
    v16bf pfr = __builtin_bit_cast(v16bf, pf);
    // P @ V using transposed V: B(n,k) = Vt[c0+..+n][j0+k]   (b128 loads)
#pragma unroll
    for (int t = 0; t < NT; ++t) {
      v16bf vf = load_frag_rowmajor(Vt + (size_t)(c0 + t * 16) * Tt + j0, Tt);
      oacc[t] = wmma_bf16(pfr, vf, oacc[t]);
    }
  }
  // normalize and store
  float lv[8];
#pragma unroll
  for (int v = 0; v < 8; ++v) lv[v] = __shfl(lrun, mbase + v, 32);
#pragma unroll
  for (int t = 0; t < NT; ++t)
#pragma unroll
    for (int v = 0; v < 8; ++v) {
      float val = oacc[t][v] / lv[v];
      obuf[(size_t)(i0 + mbase + v) * Dd + c0 + t * 16 + nloc] = f32_to_bf16(val);
    }
}

// ---------------------------------------------------------------------------
// host launcher
// ---------------------------------------------------------------------------
extern "C" void kernel_launch(void* const* d_in, const int* in_sizes, int n_in,
                              void* d_out, int out_size, void* d_ws, size_t ws_size,
                              hipStream_t stream) {
  const float* x    = (const float*)d_in[0];
  const float* W_in = (const float*)d_in[1];
  const float* W_o  = (const float*)d_in[2];
  const float* W1   = (const float*)d_in[3];
  const float* W2   = (const float*)d_in[4];
  const float* g1   = (const float*)d_in[5];
  const float* b1   = (const float*)d_in[6];
  const float* g2   = (const float*)d_in[7];
  const float* b2   = (const float*)d_in[8];
  const float* ls1  = (const float*)d_in[9];
  const float* ls2  = (const float*)d_in[10];
  float* out = (float*)d_out;

  char* w = (char*)d_ws;
  auto alloc = [&](size_t bytes) -> char* {
    char* p = w;
    w += (bytes + 255) & ~(size_t)255;
    return p;
  };
  bf16_t* hb    = (bf16_t*)alloc((size_t)Tt * Dd * 2);        // LN1(x)
  bf16_t* qkv   = (bf16_t*)alloc((size_t)Tt * LD3 * 2);       // q|k|v
  bf16_t* Vt    = (bf16_t*)alloc((size_t)Dd * Tt * 2 + 512);  // V transposed (+overrun pad)
  bf16_t* obuf  = (bf16_t*)alloc((size_t)Tt * Dd * 2);        // attn @ v
  float*  x1    = (float*) alloc((size_t)Tt * Dd * 4);        // x + ls1*o@Wout^T
  bf16_t* h2    = (bf16_t*)alloc((size_t)Tt * Dd * 2);        // LN2(x1)
  bf16_t* gbuf  = (bf16_t*)alloc((size_t)Tt * Dff * 2);       // gelu(h2@W1^T)
  bf16_t* WinB  = (bf16_t*)alloc((size_t)3 * Dd * Dd * 2);
  bf16_t* WoB   = (bf16_t*)alloc((size_t)Dd * Dd * 2);
  bf16_t* W1B   = (bf16_t*)alloc((size_t)Dff * Dd * 2);
  bf16_t* W2B   = (bf16_t*)alloc((size_t)Dd * Dff * 2);
  (void)ws_size; (void)in_sizes; (void)n_in; (void)out_size;

  auto cdiv = [](int a, int b) { return (a + b - 1) / b; };

  // 0) weights -> bf16
  cvt_f32_bf16<<<cdiv(3 * Dd * Dd, 256), 256, 0, stream>>>(W_in, WinB, 3 * Dd * Dd);
  cvt_f32_bf16<<<cdiv(Dd * Dd, 256), 256, 0, stream>>>(W_o, WoB, Dd * Dd);
  cvt_f32_bf16<<<cdiv(Dff * Dd, 256), 256, 0, stream>>>(W1, W1B, Dff * Dd);
  cvt_f32_bf16<<<cdiv(Dd * Dff, 256), 256, 0, stream>>>(W2, W2B, Dd * Dff);

  // 1) h = LN(x)
  layernorm_bf16<<<Tt / 8, 256, 0, stream>>>(x, g1, b1, hb, Tt, Dd);

  // 2) qkv = h @ W_in^T   (M=T, N=1536, K=512)
  {
    int waves = (Tt / 32) * (LD3 / 64);
    gemm_wmma<2, 4, 0><<<cdiv(waves, 8), 256, 0, stream>>>(
        hb, Dd, WinB, Dd, Tt, LD3, Dd, qkv, nullptr, LD3, nullptr, nullptr);
  }

  // 3) Vt[c][j] = V[j][c]
  transpose_bf16<<<dim3(Dd / 32, Tt / 32), 256, 0, stream>>>(qkv + 1024, LD3, Vt, Tt);

  // 4) flash attention -> obuf (bf16 [T,512])
  {
    int waves = (Tt / 16) * (Dd / 128);   // NT=8 -> 128 cols per wave
    attn_flash<8><<<cdiv(waves, 8), 256, 0, stream>>>(qkv, Vt, obuf, Tt);
  }

  // 5) x1 = x + ls1 * (obuf @ W_out^T)
  {
    int waves = (Tt / 32) * (Dd / 64);
    gemm_wmma<2, 4, 2><<<cdiv(waves, 8), 256, 0, stream>>>(
        obuf, Dd, WoB, Dd, Tt, Dd, Dd, nullptr, x1, Dd, x, ls1);
  }

  // 6) h2 = LN(x1)
  layernorm_bf16<<<Tt / 8, 256, 0, stream>>>(x1, g2, b2, h2, Tt, Dd);

  // 7) gbuf = gelu(h2 @ W1^T)   (M=T, N=2048, K=512)
  {
    int waves = (Tt / 32) * (Dff / 64);
    gemm_wmma<2, 4, 1><<<cdiv(waves, 8), 256, 0, stream>>>(
        h2, Dd, W1B, Dd, Tt, Dff, Dd, gbuf, nullptr, Dff, nullptr, nullptr);
  }

  // 8) out = x1 + ls2 * (gbuf @ W2^T)   (M=T, N=512, K=2048)
  {
    int waves = (Tt / 32) * (Dd / 64);
    gemm_wmma<2, 4, 2><<<cdiv(waves, 8), 256, 0, stream>>>(
        gbuf, Dff, W2B, Dff, Tt, Dd, Dff, nullptr, out, Dd, x1, ls2);
  }
}
